// QueryProposal_16063177687533
// MI455X (gfx1250) — matrix operand order repaired and checked by reading, not compiled, not run on hardware
//
#include <hip/hip_runtime.h>

typedef __attribute__((ext_vector_type(16))) _Float16 v16h;
typedef __attribute__((ext_vector_type(8)))  _Float16 v8h_t;
typedef __attribute__((ext_vector_type(8)))  float    v8f;
typedef __attribute__((ext_vector_type(4)))  unsigned int u32x4;
typedef __attribute__((ext_vector_type(8)))  int      i32x8;
typedef __attribute__((ext_vector_type(4)))  int      i32x4;

#define BB 8
#define CC 256
#define HH 128
#define WW 128
#define HP 130            // padded H (1-px halo)
#define WP 130            // padded W
#define HW (HH*WW)        // 16384
#define NCLS 80
#define NC1 81
#define NPAD 96
#define KK1 2304          // 9*256
#define EPSV 1e-6f
#define TOPK_N 100

// Build a 16-half fragment from two 16-byte-aligned 8-half chunks (global or LDS).
__device__ inline v16h load_frag(const _Float16* p0, const _Float16* p1) {
  v8h_t lo = *(const v8h_t*)p0;
  v8h_t hi = *(const v8h_t*)p1;
  v16h r;
#pragma unroll
  for (int i = 0; i < 8; ++i) { r[i] = lo[i]; r[8 + i] = hi[i]; }
  return r;
}

// ---- Tensor Data Mover: 2-D tile load (f16) global -> LDS -----------------
// Loads tile [tileY rows][tileX halfs] from a row-major matrix with rowStride
// halfs per row; rows packed contiguously into LDS at ldsByteOff.
__device__ inline void tdm_load_2d(const _Float16* gsrc, unsigned ldsByteOff,
                                   unsigned tileX, unsigned tileY,
                                   unsigned rowStride) {
  unsigned long long ga = (unsigned long long)(const void*)gsrc;
  u32x4 g0;
  g0[0] = 1u;                                     // count=1, user load
  g0[1] = ldsByteOff;                             // D#.lds_addr
  g0[2] = (unsigned)(ga & 0xFFFFFFFFu);           // global_addr[31:0]
  g0[3] = (unsigned)((ga >> 32) & 0x01FFFFFFu) | (2u << 30); // addr[56:32]|type=2
  i32x8 g1;
  g1[0] = 1 << 16;                                // data_size=2B; mask=0
  g1[1] = (int)((rowStride & 0xFFFFu) << 16);     // tensor_dim0[15:0]
  g1[2] = (int)((rowStride >> 16) & 0xFFFFu) | (int)((tileY & 0xFFFFu) << 16);
  g1[3] = (int)((tileX & 0xFFFFu) << 16);         // tensor_dim1 hi=0 | tile_dim0
  g1[4] = (int)(tileY & 0xFFFFu);                 // tile_dim1 | tile_dim2=0
  g1[5] = (int)rowStride;                         // tensor_dim0_stride[31:0]
  g1[6] = 0;
  g1[7] = 0;
  i32x4 z4; z4[0] = z4[1] = z4[2] = z4[3] = 0;
  i32x8 z8;
#pragma unroll
  for (int i = 0; i < 8; ++i) z8[i] = 0;
  __builtin_amdgcn_tensor_load_to_lds(g0, g1, z4, z4, z8, 0);
}

// ---------------- packing / conversion kernels ----------------

__global__ void k_pack_w1(const float* __restrict__ w1, _Float16* __restrict__ wp) {
  int t = blockIdx.x * blockDim.x + threadIdx.x;
  if (t >= CC * KK1) return;
  int n = t / KK1, kk = t - n * KK1;
  int s = kk >> 8, cin = kk & 255;
  int ky = s / 3, kx = s - ky * 3;
  wp[t] = (_Float16)w1[(((long)n * CC + cin) * 3 + ky) * 3 + kx];
}

__global__ void k_pack_w2(const float* __restrict__ w2, _Float16* __restrict__ w2p) {
  int t = blockIdx.x * blockDim.x + threadIdx.x;
  if (t >= NPAD * CC) return;
  int n = t >> 8, cin = t & 255;
  w2p[t] = (n < NC1) ? (_Float16)w2[n * CC + cin] : (_Float16)0.0f;
}

__global__ void k_fill_xh(_Float16* __restrict__ xh2) {
  long t = (long)blockIdx.x * blockDim.x + threadIdx.x;
  if (t >= (long)BB * HP * WP * CC) return;
  xh2[t] = (_Float16)0.0f;
}

__global__ void k_cvt_x(const float* __restrict__ x, _Float16* __restrict__ xh2) {
  long t = (long)blockIdx.x * blockDim.x + threadIdx.x;
  if (t >= (long)BB * HW * CC) return;
  int c = (int)(t & 255);
  int p = (int)((t >> 8) & (HW - 1));
  int b = (int)(t >> 22);
  int h = p >> 7, w = p & 127;
  xh2[(((long)b * HP + (h + 1)) * WP + (w + 1)) * CC + c] =
      (_Float16)x[((long)b * CC + c) * HW + p];
}

// ---------------- conv 3x3 implicit GEMM (WMMA f16->f32, TDM-staged B) -------
// Block: 8 waves = one image row x 64 out-channels. B slab (64x256 halfs) is
// TDM-loaded into LDS, double buffered; compute on slab s overlaps fetch s+1.

__global__ __launch_bounds__(256) void k_conv3x3_wmma(
    const _Float16* __restrict__ xh2, const _Float16* __restrict__ wp,
    const float* __restrict__ b1, _Float16* __restrict__ yh) {
  __shared__ _Float16 Bs[2][64 * CC];   // 2 x 32 KB

  const int lane = threadIdx.x & 31;
  const int wv   = threadIdx.x >> 5;
  const int hi   = lane >> 4;
  const int ln   = lane & 15;

  int blk  = blockIdx.x;
  int b    = blk >> 9;                  // 512 blocks per batch
  int rem  = blk & 511;
  int row  = rem >> 2;                  // image row
  int nblk = rem & 3;                   // 64-channel group

  const int m0   = row * WW + wv * 16;
  const int n0   = nblk * 64;
  const int wpix = wv * 16 + ln;
  const int kbA  = hi ? 8 : 0;
  const int kbB  = hi ? 8 : 0;
  const int ncol = n0 + ln;

  const bool w0 = (threadIdx.x < 32);
  const unsigned ldsBase = (unsigned)(unsigned long long)(const void*)&Bs[0][0];

  v8f acc[4];
#pragma unroll
  for (int j = 0; j < 4; ++j)
#pragma unroll
    for (int e = 0; e < 8; ++e) acc[j][e] = 0.0f;

  if (w0)
    tdm_load_2d(wp + (long)n0 * KK1, ldsBase, CC, 64, KK1);

  for (int s = 0; s < 9; ++s) {
    const int buf = s & 1;
    if (w0) {
      if (s + 1 < 9) {
        tdm_load_2d(wp + (long)n0 * KK1 + (s + 1) * CC,
                    ldsBase + (unsigned)((buf ^ 1) * (64 * CC * 2)), CC, 64, KK1);
        __builtin_amdgcn_s_wait_tensorcnt((short)1);   // slab s landed
      } else {
        __builtin_amdgcn_s_wait_tensorcnt((short)0);
      }
    }
    __syncthreads();

    const int ky = s / 3, kx = s - ky * 3;
    const _Float16* arow =
        xh2 + (((long)b * HP + (row + ky)) * WP + (wpix + kx)) * CC;
    __builtin_prefetch(arow + (long)WP * CC, 0, 1);    // next image row

#pragma unroll
    for (int c0 = 0; c0 < CC; c0 += 32) {
      const _Float16* ap = arow + c0 + kbA;
      v16h a = load_frag(ap, ap + 16);
#pragma unroll
      for (int j = 0; j < 4; ++j) {
        const int bi = (ln + 16 * j) * CC + c0 + kbB;
        v16h bf = load_frag(&Bs[buf][bi], &Bs[buf][bi + 16]);
        acc[j] = __builtin_amdgcn_wmma_f32_16x16x32_f16(
            false, a, false, bf, (short)0, acc[j], false, false);
      }
    }
    __syncthreads();
  }

  const long ybase = (long)b * HW;
#pragma unroll
  for (int j = 0; j < 4; ++j) {
    const int n = ncol + 16 * j;
    const float bias = b1[n];
    const int mb = m0 + hi * 8;
#pragma unroll
    for (int r = 0; r < 8; ++r) {
      float v = acc[j][r] + bias;
      v = fmaxf(v, 0.0f);
      yh[(ybase + mb + r) * CC + n] = (_Float16)v;
    }
  }
}

// ---------------- conv 1x1 (logits) via WMMA, TDM-staged weights ------------

__global__ __launch_bounds__(256) void k_conv1x1_wmma(
    const _Float16* __restrict__ yh, const _Float16* __restrict__ w2p,
    const float* __restrict__ b2, float* __restrict__ logits) {
  __shared__ _Float16 Ws[NPAD * CC];    // 48 KB

  const int lane = threadIdx.x & 31;
  const int wv = threadIdx.x >> 5;
  const int hi = lane >> 4;
  const int ln = lane & 15;

  int blk = blockIdx.x;
  int b = blk >> 7;
  int pixblk = blk & 127;
  const int m0 = pixblk * 128 + wv * 16;
  const int kbA = hi ? 8 : 0;
  const int kbB = hi ? 8 : 0;

  if (threadIdx.x < 32) {
    tdm_load_2d(w2p, (unsigned)(unsigned long long)(const void*)&Ws[0],
                CC, NPAD, CC);
    __builtin_amdgcn_s_wait_tensorcnt((short)0);
  }
  __syncthreads();

  const _Float16* abase = yh + ((long)b * HW + m0 + ln) * CC;

  v8f acc[6];
#pragma unroll
  for (int j = 0; j < 6; ++j)
#pragma unroll
    for (int e = 0; e < 8; ++e) acc[j][e] = 0.0f;

#pragma unroll
  for (int c0 = 0; c0 < CC; c0 += 32) {
    const _Float16* ap = abase + c0 + kbA;
    v16h a = load_frag(ap, ap + 16);
#pragma unroll
    for (int j = 0; j < 6; ++j) {
      const int bi = (16 * j + ln) * CC + c0 + kbB;
      v16h bf = load_frag(&Ws[bi], &Ws[bi + 16]);
      acc[j] = __builtin_amdgcn_wmma_f32_16x16x32_f16(
          false, a, false, bf, (short)0, acc[j], false, false);
    }
  }

#pragma unroll
  for (int j = 0; j < 6; ++j) {
    const int n = 16 * j + ln;
    if (n < NC1) {
      const float bias = b2[n];
      const int mb = m0 + hi * 8;
#pragma unroll
      for (int r = 0; r < 8; ++r)
        logits[((long)b * NC1 + n) * HW + mb + r] = acc[j][r] + bias;
    }
  }
}

// ---------------- softmax + argmax (winning class prob per pixel) -----------

__global__ void k_softmax_argmax(const float* __restrict__ logits,
                                 int* __restrict__ cls, float* __restrict__ pwin) {
  int t = blockIdx.x * blockDim.x + threadIdx.x;
  if (t >= BB * HW) return;
  int b = t >> 14, p = t & (HW - 1);
  const float* lp = logits + (long)b * NC1 * HW + p;
  float mx = -1e30f;
  for (int n = 0; n < NC1; ++n) mx = fmaxf(mx, lp[(long)n * HW]);
  float sum = 0.0f, bestl = -1e30f;
  int best = 0;
  for (int n = 0; n < NC1; ++n) {
    float l = lp[(long)n * HW];
    sum += __expf(l - mx);
    if (n < NCLS && l > bestl) { bestl = l; best = n; }
  }
  cls[t] = best;
  pwin[t] = __expf(bestl - mx) / sum;
}

// ---------------- same-class 3x3 local-max boost -> score map ----------------

__global__ void k_nms_score(const int* __restrict__ cls,
                            const float* __restrict__ pwin,
                            float* __restrict__ score) {
  int t = blockIdx.x * blockDim.x + threadIdx.x;
  if (t >= BB * HW) return;
  int b = t >> 14, p = t & (HW - 1);
  int h = p >> 7, w = p & 127;
  int c = cls[t];
  float pv = pwin[t];
  bool ismax = (pv >= EPSV);
  for (int dy = -1; dy <= 1; ++dy)
    for (int dx = -1; dx <= 1; ++dx) {
      if (dy == 0 && dx == 0) continue;
      int hh = h + dy, ww = w + dx;
      if ((unsigned)hh < (unsigned)HH && (unsigned)ww < (unsigned)WW) {
        int q = (b << 14) + (hh << 7) + ww;
        float nv = (cls[q] == c) ? pwin[q] : 0.0f;
        if (pv < nv) ismax = false;
      }
    }
  score[t] = pv + (ismax ? 1.0f : 0.0f);
}

// ---------------- top-100 per batch (iterative argmax over LDS copy) ---------

__global__ __launch_bounds__(1024) void k_topk(const float* __restrict__ score,
                                               int* __restrict__ idxo) {
  __shared__ float sv[HW];        // 64 KB
  __shared__ float rv[1024];
  __shared__ int   ri[1024];
  const int b = blockIdx.x, t = threadIdx.x;
  for (int i = t; i < HW; i += 1024) sv[i] = score[(long)b * HW + i];
  __syncthreads();
  for (int k = 0; k < TOPK_N; ++k) {
    float best = -1e30f; int bi = HW;
    for (int i = t; i < HW; i += 1024) {
      float v = sv[i];
      if (v > best) { best = v; bi = i; }   // ascending stride => first max wins
    }
    rv[t] = best; ri[t] = bi;
    __syncthreads();
    for (int off = 512; off > 0; off >>= 1) {
      if (t < off) {
        if (rv[t + off] > rv[t] ||
            (rv[t + off] == rv[t] && ri[t + off] < ri[t])) {
          rv[t] = rv[t + off]; ri[t] = ri[t + off];
        }
      }
      __syncthreads();
    }
    if (t == 0) { idxo[b * TOPK_N + k] = ri[0]; sv[ri[0]] = -1e30f; }
    __syncthreads();
  }
}

// ---------------- gather proposals + pos embeddings ----------------

__global__ void k_gather(const float* __restrict__ x, const float* __restrict__ pe,
                         const int* __restrict__ idxo,
                         float* __restrict__ outProp, float* __restrict__ outPos) {
  int t = blockIdx.x * blockDim.x + threadIdx.x;
  if (t >= BB * CC * TOPK_N) return;
  int j = t % TOPK_N;
  int c = (t / TOPK_N) % CC;
  int b = t / (TOPK_N * CC);
  int p = idxo[b * TOPK_N + j];
  outProp[t] = x[((long)b * CC + c) * HW + p];
  outPos[t]  = pe[(long)c * HW + p];
}

// ---------------- launch ----------------

extern "C" void kernel_launch(void* const* d_in, const int* in_sizes, int n_in,
                              void* d_out, int out_size, void* d_ws, size_t ws_size,
                              hipStream_t stream) {
  (void)in_sizes; (void)n_in; (void)out_size; (void)ws_size;
  const float* x  = (const float*)d_in[0];
  const float* pe = (const float*)d_in[1];
  const float* w1 = (const float*)d_in[2];
  const float* b1 = (const float*)d_in[3];
  const float* w2 = (const float*)d_in[4];
  const float* b2 = (const float*)d_in[5];

  float* out      = (float*)d_out;
  float* outProp  = out;                               // [8,256,100]
  float* outPos   = out + (long)BB * CC * TOPK_N;      // [8,256,100]
  float* logits   = out + (long)2 * BB * CC * TOPK_N;  // [8,81,128,128]

  char* ws = (char*)d_ws;
  _Float16* xh2 = (_Float16*)(ws);                     // 69,222,400 B (halo)
  _Float16* yh  = (_Float16*)(ws + 69222400);          // 67,108,864 B
  _Float16* wp  = (_Float16*)(ws + 136331264);         //  1,179,648 B
  _Float16* w2p = (_Float16*)(ws + 137510912);         //     49,152 B
  int*   cls  = (int*)  (ws + 137560064);              //    524,288 B
  float* pwin = (float*)(ws + 138084352);              //    524,288 B
  float* scr  = (float*)(ws + 138608640);              //    524,288 B
  int*   idxo = (int*)  (ws + 139132928);              //      3,200 B

  const long NXH = (long)BB * HP * WP * CC;            // 34,611,200 halo elems

  k_pack_w1<<<(CC * KK1 + 255) / 256, 256, 0, stream>>>(w1, wp);
  k_pack_w2<<<(NPAD * CC + 255) / 256, 256, 0, stream>>>(w2, w2p);
  k_fill_xh<<<(int)((NXH + 255) / 256), 256, 0, stream>>>(xh2);
  k_cvt_x<<<(int)(((long)BB * HW * CC + 255) / 256), 256, 0, stream>>>(x, xh2);
  k_conv3x3_wmma<<<BB * 512, 256, 0, stream>>>(xh2, wp, b1, yh);
  k_conv1x1_wmma<<<BB * 128, 256, 0, stream>>>(yh, w2p, b2, logits);
  k_softmax_argmax<<<(BB * HW + 255) / 256, 256, 0, stream>>>(logits, cls, pwin);
  k_nms_score<<<(BB * HW + 255) / 256, 256, 0, stream>>>(cls, pwin, scr);
  k_topk<<<BB, 1024, 0, stream>>>(scr, idxo);
  k_gather<<<(BB * CC * TOPK_N + 255) / 256, 256, 0, stream>>>(x, pe, idxo, outProp, outPos);
}
